// MultiView_4956392259845
// MI455X (gfx1250) — compile-verified, matrix operand order checked by reading, not run on hardware
//
#include <hip/hip_runtime.h>

typedef unsigned short u16;
typedef __attribute__((ext_vector_type(16))) __bf16 v16bf;
typedef __attribute__((ext_vector_type(16))) u16   v16u;
typedef __attribute__((ext_vector_type(8)))  u16   u16x8;
typedef __attribute__((ext_vector_type(8)))  float v8f;
typedef __attribute__((ext_vector_type(4)))  unsigned int u32x4;
typedef __attribute__((ext_vector_type(8)))  int   i32x8;
typedef __attribute__((ext_vector_type(4)))  int   i32x4;

__device__ __forceinline__ u16 f2bf(float f) {
    unsigned int u = __builtin_bit_cast(unsigned int, f);
    unsigned int r = u + 0x7FFFu + ((u >> 16) & 1u);   // round-to-nearest-even
    return (u16)(r >> 16);
}
__device__ __forceinline__ float bf2f(u16 h) {
    unsigned int u = ((unsigned int)h) << 16;
    return __builtin_bit_cast(float, u);
}

// ---------------------------------------------------------------------------
// Implicit-GEMM conv, channels-last (N,D,H,W,C), weights [Co][tap][ci] bf16.
// Block (256 thr = 8 waves) = one 16-row Co tile x 256 output positions.
// The 16x32 bf16 weight chunk is staged into double-buffered LDS by the
// Tensor Data Mover (tensor_load_to_lds, TENSORcnt-tracked), fragments are
// read back with ds_load_b128; B fragments are two aligned b128 global loads
// per subtile (ci is pow2 and a multiple of 8 on all non-stem convs).
// Each wave: 2 position subtiles -> 2 x v_wmma_f32_16x16x32_bf16 per chunk.
// Epilogue: folded BN scale/bias, optional residual, ReLU, one b128 store.
// ---------------------------------------------------------------------------
__global__ void __launch_bounds__(256)
conv_wmma_kernel(const u16* __restrict__ X, const u16* __restrict__ Wt,
                 const float* __restrict__ SB, const u16* __restrict__ RES,
                 u16* __restrict__ Y,
                 int Ci, int ciLog2, int Dd, int Hh, int Ww,
                 int Co, int KHWt, int KW,
                 int pd, int ph, int pw, int relu,
                 int K, int Npos, int tilesM)
{
    __shared__ u16 lds[2][512];

    int lane = threadIdx.x & 31;
    int half = lane >> 4;
    int m    = lane & 15;
    int wid  = threadIdx.x >> 5;

    int tm   = blockIdx.x % tilesM;
    int tg   = blockIdx.x / tilesM;
    int base = tg * 256 + wid * 32;

    // Visible (but never executed) store: the LDS tile is written by the TDM,
    // which the compiler cannot see; without this it would fold the ds_loads
    // of a "never-stored" internal LDS global to poison.
    if (Npos < 0) lds[0][0] = (u16)Npos;

    int HW = Hh * Ww, DHW = Dd * HW;

    bool valid[2];
    int pos[2], p0[2], od[2], oh[2], ow[2];
#pragma unroll
    for (int st = 0; st < 2; ++st) {
        int pb    = base + st * 16;
        valid[st] = (pb < Npos);                   // Npos is a multiple of 16
        int p     = (valid[st] ? pb : 0) + m;
        pos[st]   = p;
        int nb = p / DHW;  int r1 = p - nb * DHW;
        od[st] = r1 / HW;  int r2 = r1 - od[st] * HW;
        oh[st] = r2 / Ww;  ow[st] = r2 - oh[st] * Ww;
        p0[st] = nb * DHW;
    }

    int Cm1 = Ci - 1;
    v8f acc0 = {}, acc1 = {};

    for (int k0 = 0, pbuf = 0; k0 < K; k0 += 32, pbuf ^= 1) {
        if (wid == 0) {
            // TDM: stage 16 rows x 32 bf16 (row stride K) into LDS buffer pbuf.
            // D# group0: count=1 | lds_addr[63:32] | global_addr[120:64] | type=2
            unsigned long long ga =
                (unsigned long long)(Wt + (size_t)(tm * 16) * K + k0);
            u32x4 g0;
            g0.x = 1u;                               // count = 1 valid descriptor
            g0.y = (unsigned)(pbuf * 1024);          // LDS byte address
            g0.z = (unsigned)ga;
            g0.w = (unsigned)(ga >> 32) | (2u << 30);// addr[56:32] | type=2
            // D# group1: data_size=2B; tensor_dim0=K-k0 (tail reads -> 0);
            // tensor_dim1=16; tile_dim0=32; tile_dim1=16; dim0_stride=K.
            unsigned td0 = (unsigned)(K - k0);
            i32x8 g1;
            g1[0] = 0x00010000;                      // data_size code 1 (2 bytes)
            g1[1] = (int)((td0 & 0xFFFFu) << 16);    // tensor_dim0[15:0]
            g1[2] = (int)((td0 >> 16) | (16u << 16));// tensor_dim0[31:16] | dim1
            g1[3] = (int)(32u << 16);                // tile_dim0 = 32
            g1[4] = 16;                              // tile_dim1 = 16
            g1[5] = (int)(unsigned)K;                // tensor_dim0_stride lo
            g1[6] = 0;                               // stride hi / dim1_stride
            g1[7] = 0;
            i32x4 z4 = {};
            i32x8 z8 = {};
            __builtin_amdgcn_tensor_load_to_lds(g0, g1, z4, z4, z8, 0);
            __builtin_amdgcn_s_wait_tensorcnt(0);
        }
        __syncthreads();
        // Compiler memory barrier: LDS contents were changed by the TDM engine
        // behind the optimizer's back — force the ds_loads to happen.
        asm volatile("" ::: "memory");

        // A fragment: two ds_load_b128 per lane
        u16x8 a0 = *(const u16x8*)&lds[pbuf][m * 32 + (half << 3)];
        u16x8 a1 = *(const u16x8*)&lds[pbuf][m * 32 + 16 + (half << 3)];
        v16u  au = __builtin_shufflevector(a0, a1, 0, 1, 2, 3, 4, 5, 6, 7,
                                           8, 9, 10, 11, 12, 13, 14, 15);
        v16bf av = __builtin_bit_cast(v16bf, au);

#pragma unroll
        for (int st = 0; st < 2; ++st) {
            if (!valid[st]) continue;              // wave-uniform
            v16u bu;
            if (Cm1) {  // vector path: Ci pow2, multiple of 8
                u16x8 bg0 = {}, bg1 = {};
#pragma unroll
                for (int g = 0; g < 2; ++g) {
                    int kb = k0 + (g << 4) + (half << 3);
                    u16x8 bv = {};
                    if (kb < K) {
                        int tap = kb >> ciLog2;
                        int ci0 = kb & Cm1;
                        int kd = tap / KHWt; int tr = tap - kd * KHWt;
                        int kh = tr / KW;    int kw = tr - kh * KW;
                        int id = od[st] - pd + kd;
                        int ih = oh[st] - ph + kh;
                        int iw = ow[st] - pw + kw;
                        if ((unsigned)id < (unsigned)Dd &&
                            (unsigned)ih < (unsigned)Hh &&
                            (unsigned)iw < (unsigned)Ww)
                            bv = *(const u16x8*)(X +
                                 ((size_t)(p0[st] + id * HW + ih * Ww + iw) * Ci + ci0));
                    }
                    if (g == 0) bg0 = bv; else bg1 = bv;
                }
                bu = __builtin_shufflevector(bg0, bg1, 0, 1, 2, 3, 4, 5, 6, 7,
                                             8, 9, 10, 11, 12, 13, 14, 15);
            } else {    // Ci == 1 (7x7 stem): scalar gather, tap == k
#pragma unroll
                for (int e = 0; e < 16; ++e) {
                    int k = k0 + (e & 7) + (half << 3) + ((e >> 3) << 4);
                    u16 bval = 0;
                    if (k < K) {
                        int kd = k / KHWt; int tr = k - kd * KHWt;
                        int kh = tr / KW;  int kw = tr - kh * KW;
                        int id = od[st] - pd + kd;
                        int ih = oh[st] - ph + kh;
                        int iw = ow[st] - pw + kw;
                        if ((unsigned)id < (unsigned)Dd &&
                            (unsigned)ih < (unsigned)Hh &&
                            (unsigned)iw < (unsigned)Ww)
                            bval = X[(size_t)(p0[st] + id * HW + ih * Ww + iw)];
                    }
                    bu[e] = bval;
                }
            }
            v16bf bv16 = __builtin_bit_cast(v16bf, bu);
            if (st == 0)
                acc0 = __builtin_amdgcn_wmma_f32_16x16x32_bf16(
                           false, av, false, bv16, (short)0, acc0, false, false);
            else
                acc1 = __builtin_amdgcn_wmma_f32_16x16x32_bf16(
                           false, av, false, bv16, (short)0, acc1, false, false);
        }
    }

    // Epilogue: 8 contiguous co per lane -> single 16B store (channels-last)
    int cob = tm * 16 + (half << 3);
#pragma unroll
    for (int st = 0; st < 2; ++st) {
        if (!valid[st]) continue;
        v8f a = st ? acc1 : acc0;
        size_t ob = (size_t)pos[st] * Co + cob;
        float vv[8];
#pragma unroll
        for (int r = 0; r < 8; ++r)
            vv[r] = a[r] * SB[cob + r] + SB[Co + cob + r];
        if (RES) {
            u16x8 rv = *(const u16x8*)(RES + ob);
#pragma unroll
            for (int r = 0; r < 8; ++r) vv[r] += bf2f(rv[r]);
        }
        u16x8 outv;
#pragma unroll
        for (int r = 0; r < 8; ++r) {
            float v = vv[r];
            if (relu && v < 0.f) v = 0.f;
            outv[r] = f2bf(v);
        }
        *(u16x8*)(Y + ob) = outv;
    }
}

// ---------------------------------------------------------------------------
// Support kernels (all channels-last)
// ---------------------------------------------------------------------------
__global__ void f32_to_bf16_k(const float* __restrict__ x, u16* __restrict__ y, int n) {
    int t = blockIdx.x * blockDim.x + threadIdx.x;
    if (t < n) y[t] = f2bf(x[t]);
}

__global__ void bn_fold_k(const float* g, const float* b, const float* m,
                          const float* v, float* sb, int C) {
    int t = threadIdx.x;
    if (t < C) {
        float s = g[t] / sqrtf(v[t] + 1e-3f);
        sb[t]     = s;
        sb[C + t] = b[t] - m[t] * s;
    }
}

// (Co,Ci,KD,KH,KW) fp32 -> bf16 [Co][tap][ci], tap = (kd*KH+kh)*KW+kw
__global__ void prep_weight_k(const float* __restrict__ w, u16* __restrict__ o,
                              int Ci, int taps, int n) {
    int t = blockIdx.x * blockDim.x + threadIdx.x;
    if (t >= n) return;
    int tap = t % taps;
    int ci  = (t / taps) % Ci;
    int co  = t / (taps * Ci);
    o[(size_t)co * taps * Ci + (size_t)tap * Ci + ci] = f2bf(w[t]);
}

// x (B, S,S,S, C) -> v (S*B, S, S, C), slicing axis ax in {2,3,4} (NCDHW axes)
__global__ void extract_view_k(const u16* __restrict__ x, u16* __restrict__ v,
                               int Bn, int C, int S, int ax, int n) {
    int t = blockIdx.x * blockDim.x + threadIdx.x;
    if (t >= n) return;
    int c = t % C;  int r = t / C;
    int j = r % S;  r /= S;
    int i = r % S;  r /= S;
    int b = r % Bn;
    int s = r / Bn;
    int d, h, w;
    if      (ax == 2) { d = s; h = i; w = j; }
    else if (ax == 3) { d = i; h = s; w = j; }
    else              { d = i; h = j; w = s; }
    v[t] = x[((((size_t)b * S + d) * S + h) * S + w) * C + c];
}

// y (S*B, S, S, Co) -> acc (B, S,S,S, Co) float, inverse view permutation
__global__ void accumulate_view_k(const u16* __restrict__ y, float* __restrict__ acc,
                                  int Bn, int Co, int S, int ax, int first, int n) {
    int t = blockIdx.x * blockDim.x + threadIdx.x;
    if (t >= n) return;
    int c  = t % Co; int r = t / Co;
    int vj = r % S;  r /= S;
    int ui = r % S;  r /= S;
    int b  = r % Bn;
    int s  = r / Bn;
    int d, h, w;
    if      (ax == 2) { d = s;  h = ui; w = vj; }
    else if (ax == 3) { d = ui; h = s;  w = vj; }
    else              { d = ui; h = vj; w = s;  }
    size_t o = ((((size_t)b * S + d) * S + h) * S + w) * Co + c;
    float val = bf2f(y[t]);
    if (first) acc[o] = val; else acc[o] += val;
}

__global__ void scale_bf16_k(const float* __restrict__ a, u16* __restrict__ o, int n) {
    int t = blockIdx.x * blockDim.x + threadIdx.x;
    if (t < n) o[t] = f2bf(a[t] * (1.0f / 3.0f));
}

// 2x2x2 maxpool, channels-last input (N, 2Do,2Ho,2Wo, C).
// writef=0: bf16 channels-last out. writef=1: fp32 NCDHW out (final d_out).
__global__ void maxpool3d_k(const u16* __restrict__ x, u16* __restrict__ yb,
                            float* __restrict__ yf, int writef,
                            int C, int Do, int Ho, int Wo, int n) {
    int t = blockIdx.x * blockDim.x + threadIdx.x;
    if (t >= n) return;
    int c = t % C;  int r = t / C;
    int w = r % Wo; r /= Wo;
    int h = r % Ho; r /= Ho;
    int d = r % Do;
    int b = r / Do;
    int Di = 2 * Do, Hi = 2 * Ho, Wi = 2 * Wo;
    float mx = -3.0e38f;
#pragma unroll
    for (int dz = 0; dz < 2; ++dz)
#pragma unroll
        for (int dy = 0; dy < 2; ++dy)
#pragma unroll
            for (int dx = 0; dx < 2; ++dx) {
                size_t idx = ((((size_t)b * Di + 2 * d + dz) * Hi + 2 * h + dy) * Wi
                              + 2 * w + dx) * C + c;
                float vv = bf2f(x[idx]);
                mx = vv > mx ? vv : mx;
            }
    if (writef)
        yf[((((size_t)b * C + c) * Do + d) * Ho + h) * Wo + w] = mx;
    else
        yb[t] = f2bf(mx);
}

// ---------------------------------------------------------------------------
// Host orchestration
// ---------------------------------------------------------------------------
extern "C" void kernel_launch(void* const* d_in, const int* in_sizes, int n_in,
                              void* d_out, int out_size, void* d_ws, size_t ws_size,
                              hipStream_t stream) {
    (void)in_sizes; (void)n_in; (void)out_size; (void)ws_size;
    char* wsb = (char*)d_ws;
    size_t cur = 0;
    auto alloc = [&](size_t bytes) -> void* {
        void* p = wsb + cur;
        cur += (bytes + 255) & ~(size_t)255;
        return p;
    };

    const size_t BIG = 16384000;                 // max activation elems (160,80,80,16)
    u16*   A   = (u16*)alloc(BIG * 2);
    u16*   Bb  = (u16*)alloc(BIG * 2);
    u16*   Cc  = (u16*)alloc(BIG * 2);
    u16*   Dc  = (u16*)alloc(BIG * 2);           // current level input
    u16*   Vv  = (u16*)alloc((size_t)2200000 * 2);
    float* ACC = (float*)alloc(BIG * 4);

    // d_in flat order (setup_inputs insertion order): x, then param leaves.
    // Every conv weight at index i has its BN g/b/m/v at i+1..i+4.
    struct WInfo { int idx, Co, Ci, taps; };     // K = taps*Ci
    static const WInfo WL[23] = {
        {  1,  16,   1, 49},                                      // 0  l0 7x7 stem
        {  6,  16,  16, 27},                                      // 1  c3d0
        { 11,  32,  16, 9}, { 16,  32,  32, 9}, { 21,  32,  16, 1},  // 2-4  l1a
        { 26,  32,  32, 9}, { 31,  32,  32, 9}, { 36,  32,  32, 1},  // 5-7  l1b
        { 41,  32,  32, 27},                                      // 8  c3d1
        { 46,  64,  32, 9}, { 51,  64,  64, 9}, { 56,  64,  32, 1},  // 9-11 l2a
        { 61,  64,  64, 9}, { 66,  64,  64, 9}, { 71,  64,  64, 1},  // 12-14 l2b
        { 76,  64,  64, 27},                                      // 15 c3d2
        { 81, 128,  64, 9}, { 86, 128, 128, 9}, { 91, 128,  64, 1},  // 16-18 l3a
        { 96, 128, 128, 9}, {101, 128, 128, 9}, {106, 128, 128, 1},  // 19-21 l3b
        {111, 128, 128, 27}                                       // 22 c3d3
    };

    u16*   WB[23];
    float* SBp[23];
    for (int i = 0; i < 23; ++i) {
        int Co = WL[i].Co, Ci = WL[i].Ci, taps = WL[i].taps, idx = WL[i].idx;
        int nw = Co * Ci * taps;
        WB[i]  = (u16*)alloc((size_t)nw * 2);
        SBp[i] = (float*)alloc((size_t)Co * 2 * 4);
        prep_weight_k<<<(nw + 255) / 256, 256, 0, stream>>>(
            (const float*)d_in[idx], WB[i], Ci, taps, nw);
        bn_fold_k<<<1, 128, 0, stream>>>(
            (const float*)d_in[idx + 1], (const float*)d_in[idx + 2],
            (const float*)d_in[idx + 3], (const float*)d_in[idx + 4],
            SBp[i], Co);
    }

    auto conv = [&](const u16* Xp, int wi, const u16* res, u16* Yp,
                    int N, int Dd, int Hh, int Ww,
                    int KD, int KH, int KW, int pd, int ph, int pw, int relu) {
        int Co = WL[wi].Co, Ci = WL[wi].Ci;
        int K  = WL[wi].taps * Ci;
        int ciLog2 = 0; while ((1 << ciLog2) < Ci) ++ciLog2;
        int npos   = N * Dd * Hh * Ww;            // multiple of 16 in this net
        int tilesM = Co / 16;
        int gx     = (npos + 255) / 256;
        (void)KD;
        conv_wmma_kernel<<<gx * tilesM, 256, 0, stream>>>(
            Xp, WB[wi], SBp[wi], res, Yp,
            Ci, ciLog2, Dd, Hh, Ww, Co, KH * KW, KW,
            pd, ph, pw, relu, K, npos, tilesM);
    };

    // Conv-BN-ReLU -> Conv-BN (+ 1x1 Conv-BN shortcut) -> ReLU
    auto cbrcb = [&](const u16* z, int N, int S2, int wbase,
                     u16* t1, u16* sc, u16* out) {
        conv(z,  wbase + 0, nullptr, t1,  N, 1, S2, S2, 1, 3, 3, 0, 1, 1, 1);
        conv(z,  wbase + 2, nullptr, sc,  N, 1, S2, S2, 1, 1, 1, 0, 0, 0, 0);
        conv(t1, wbase + 1, sc,      out, N, 1, S2, S2, 1, 3, 3, 0, 1, 1, 1);
    };

    // x (2,1,80,80,80) fp32: C==1 so NCDHW == channels-last; convert to bf16
    int nX = 2 * 80 * 80 * 80;
    f32_to_bf16_k<<<(nX + 255) / 256, 256, 0, stream>>>((const float*)d_in[0], Dc, nX);

    struct Lv { int cbA, cbB, c3d, Ci, Co; };
    static const Lv LV[4] = {
        { -1, -1,  1,   1,  16 },   // level 0: 7x7 stem (wi 0) + c3d0
        {  2,  5,  8,  16,  32 },
        {  9, 12, 15,  32,  64 },
        { 16, 19, 22,  64, 128 }
    };

    int S = 80;
    for (int lv = 0; lv < 4; ++lv) {
        int Ci = LV[lv].Ci, Co = LV[lv].Co;
        for (int ax = 2; ax <= 4; ++ax) {
            int nV = S * 2 * Ci * S * S;
            extract_view_k<<<(nV + 255) / 256, 256, 0, stream>>>(
                Dc, Vv, 2, Ci, S, ax, nV);
            u16* y;
            if (lv == 0) {
                conv(Vv, 0, nullptr, A, S * 2, 1, S, S, 1, 7, 7, 0, 3, 3, 1);
                y = A;
            } else {
                cbrcb(Vv, S * 2, S, LV[lv].cbA, A, Bb, Cc);
                cbrcb(Cc, S * 2, S, LV[lv].cbB, A, Bb, Cc);
                y = Cc;
            }
            int nY = S * 2 * Co * S * S;
            accumulate_view_k<<<(nY + 255) / 256, 256, 0, stream>>>(
                y, ACC, 2, Co, S, ax, (ax == 2) ? 1 : 0, nY);
        }
        int nacc = 2 * Co * S * S * S;
        scale_bf16_k<<<(nacc + 255) / 256, 256, 0, stream>>>(ACC, A, nacc);
        conv(A, LV[lv].c3d, nullptr, Bb, 2, S, S, S, 3, 3, 3, 1, 1, 1, 1);
        int So = S / 2;
        int np = 2 * Co * So * So * So;
        maxpool3d_k<<<(np + 255) / 256, 256, 0, stream>>>(
            Bb, Dc, (float*)d_out, (lv == 3) ? 1 : 0, Co, So, So, So, np);
        S = So;
    }
}
